// VectorQuantizerEMA_57655640981631
// MI455X (gfx1250) — compile-verified, matrix operand order checked by reading, not run on hardware
//
#include <hip/hip_runtime.h>
#include <hip/hip_bf16.h>
#include <math.h>

// ---------------------------------------------------------------------------
// VQ-VAE EMA vector quantizer for MI455X (gfx1250, wave32, WMMA).
// B=16, T=4096, D=64, K=1024.  N = 65536 tokens.
//
// Heavy op: argmin_k ( ||w_k||^2 - 2 z.w_k ) -- 65536x1024x64 GEMM via
// v_wmma_f32_16x16x32_f16 with an f16 hi/lo split of both operands
// (3 WMMAs per 16x16 tile per 32-wide K slab => near-f32 dot products).
// Codebook is staged into LDS with GLOBAL_LOAD_ASYNC_TO_LDS_B128 (ASYNCcnt),
// double-buffered so the async DMA of chunk c+1 overlaps WMMAs on chunk c.
// ||w||^2 values for a chunk are preloaded into registers at chunk start so
// no synchronous global access sits inside the k-tile loop.
// ---------------------------------------------------------------------------

#define NTOK   65536
#define DDIM   64
#define KCODE  1024
#define DECAYC 0.99f
#define ONE_M_DECAY 0.01f
#define EPSC   1e-05f
#define COMMIT 0.25f

typedef _Float16 f16_t;
typedef __attribute__((ext_vector_type(16))) _Float16 v16h;
typedef __attribute__((ext_vector_type(8)))  _Float16 v8h;
typedef __attribute__((ext_vector_type(8)))  float    v8f;

// ---- workspace layout (bytes, all offsets 256B aligned) -------------------
#define WS_IDX       0                      // int[65536]            262144 B
#define WS_NBATCH    262144                 // float[1024]             4096 B
#define WS_EMBSUM    266240                 // float[65536]          262144 B
#define WS_LOSS      528384                 // float[1]  (pad 256)      256 B
#define WS_WHI       528640                 // f16[65536]            131072 B
#define WS_WLO       659712                 // f16[65536]            131072 B
#define WS_WSQ       790784                 // float[1024]             4096 B

// ---- out layout (floats, tuple concatenated flat) -------------------------
#define OUT_QUANT    0           // 4194304
#define OUT_LOSS     4194304     // 1
#define OUT_PERP     4194305     // 1
#define OUT_NCS      4194306     // 1024
#define OUT_NES      4195330     // 65536
#define OUT_NW       4260866     // 65536

// LDS staging: chunk of 64 codes, rows padded to 136 f16 (272 B = 68 dwords
// -> bank index 4L for lane L, conflict-free b128 fragment reads).
#define CHUNK_CODES  64
#define ROW_PAD      136
#define NCHUNK       (KCODE / CHUNK_CODES)   // 16

// ---------------------------------------------------------------------------
__global__ void k_zero(float* p, int n) {
    int i = blockIdx.x * blockDim.x + threadIdx.x;
    if (i < n) p[i] = 0.0f;
}

// One block (64 threads) per codebook row: pack f32 weight into f16 hi/lo and
// compute ||w_k||^2.
__global__ void k_prep(const float* __restrict__ w,
                       f16_t* __restrict__ whi, f16_t* __restrict__ wlo,
                       float* __restrict__ wsq) {
    const int k = blockIdx.x;
    const int d = threadIdx.x;
    float x = w[k * DDIM + d];
    _Float16 h = (_Float16)x;
    whi[k * DDIM + d] = h;
    wlo[k * DDIM + d] = (_Float16)(x - (float)h);
    __shared__ float red[DDIM];
    red[d] = x * x;
    __syncthreads();
    if (d == 0) {
        float s = 0.0f;
        #pragma unroll
        for (int i = 0; i < DDIM; ++i) s += red[i];
        wsq[k] = s;
    }
}

// Issue the async DMA of one 64-code chunk (hi+lo planes) into an LDS buffer.
// 256 threads x 4 transfers of 16B; row layout [hi(64) | lo(64) | pad(8)].
__device__ __forceinline__ void issue_chunk_copy(f16_t* dstbuf,
                                                 const f16_t* __restrict__ whi,
                                                 const f16_t* __restrict__ wlo,
                                                 int codeBase, int tid) {
    #pragma unroll
    for (int it = 0; it < 4; ++it) {
        const int q    = it * 256 + tid;          // 0..1023
        const int code = q >> 4;                  // 0..63
        const int seg  = q & 15;                  // 16B segment within row
        const f16_t* gsrc;
        int dlocal;
        if (seg < 8) { gsrc = whi + (size_t)(codeBase + code) * DDIM + seg * 8;
                       dlocal = seg * 8; }
        else         { gsrc = wlo + (size_t)(codeBase + code) * DDIM + (seg - 8) * 8;
                       dlocal = 64 + (seg - 8) * 8; }
        unsigned ldsAddr =
            (unsigned)(uintptr_t)(dstbuf + (size_t)code * ROW_PAD + dlocal);
        asm volatile("global_load_async_to_lds_b128 %0, %1, off"
                     :: "v"(ldsAddr), "v"(gsrc) : "memory");
    }
}

// WMMA argmin kernel: 256 threads = 8 waves; each wave owns 16 tokens; the
// workgroup sweeps the codebook in 16 double-buffered LDS chunks of 64 codes.
__global__ void __launch_bounds__(256)
k_argmin(const float* __restrict__ z,
         const f16_t* __restrict__ whi, const f16_t* __restrict__ wlo,
         const float* __restrict__ wsq, int* __restrict__ idxOut) {
    __shared__ f16_t smem[2][CHUNK_CODES * ROW_PAD];   // 2 x 17408 B

    const int tid     = threadIdx.x;
    const int lane    = tid & 31;
    const int wave    = tid >> 5;
    const int tokBase = blockIdx.x * 128 + wave * 16;
    const int row     = lane & 15;   // token row (A) / code col (B)
    const int half    = lane >> 4;

    // ---- A fragments: token rows, f32 -> f16 hi/lo split ------------------
    v16h a_hi[2], a_lo[2];
    const float* zr = z + (size_t)(tokBase + row) * DDIM;
    #pragma unroll
    for (int b = 0; b < 2; ++b) {
        const int dblk = b * 32;
        #pragma unroll
        for (int e = 0; e < 16; ++e) {
            const int d = dblk + (e < 8 ? half * 8 + e
                                        : 16 + half * 8 + (e - 8));
            float x = zr[d];
            _Float16 h = (_Float16)x;
            a_hi[b][e] = h;
            a_lo[b][e] = (_Float16)(x - (float)h);
        }
    }

    float minv[8];
    int   mini[8];
    #pragma unroll
    for (int j = 0; j < 8; ++j) { minv[j] = 3.4e38f; mini[j] = 0; }

    const int col = row;

    // prime the pipeline: chunk 0 -> buffer 0
    issue_chunk_copy(&smem[0][0], whi, wlo, 0, tid);

    for (int c = 0; c < NCHUNK; ++c) {
        asm volatile("s_wait_asynccnt 0x0" ::: "memory");
        __syncthreads();                       // chunk c resident in buf c&1
        if (c + 1 < NCHUNK)                    // overlap DMA of next chunk
            issue_chunk_copy(&smem[(c + 1) & 1][0], whi, wlo,
                             (c + 1) * CHUNK_CODES, tid);

        // preload this chunk's ||w||^2 values (one per k-tile for this lane):
        // loads issue before the first ds_load/WMMA and hide behind them.
        float wq[4];
        #pragma unroll
        for (int ktL = 0; ktL < 4; ++ktL)
            wq[ktL] = wsq[c * CHUNK_CODES + ktL * 16 + col];

        const f16_t* bufp = &smem[c & 1][0];
        #pragma unroll
        for (int ktL = 0; ktL < 4; ++ktL) {
            const int lc   = ktL * 16 + col;
            const int code = c * CHUNK_CODES + lc;
            const f16_t* rowp = bufp + (size_t)lc * ROW_PAD;
            v8f acc = {};
            #pragma unroll
            for (int b = 0; b < 2; ++b) {
                const int dblk = b * 32;
                v8h h0 = *(const v8h*)(rowp + dblk + half * 8);
                v8h h1 = *(const v8h*)(rowp + dblk + 16 + half * 8);
                v8h l0 = *(const v8h*)(rowp + 64 + dblk + half * 8);
                v8h l1 = *(const v8h*)(rowp + 64 + dblk + 16 + half * 8);
                v16h b_hi = __builtin_shufflevector(h0, h1, 0,1,2,3,4,5,6,7,
                                                            8,9,10,11,12,13,14,15);
                v16h b_lo = __builtin_shufflevector(l0, l1, 0,1,2,3,4,5,6,7,
                                                            8,9,10,11,12,13,14,15);
                // dot(z, w) ~= hi*hi + lo*hi + hi*lo (f32 accumulate)
                acc = __builtin_amdgcn_wmma_f32_16x16x32_f16(
                          false, a_hi[b], false, b_hi, (short)0, acc, false, false);
                acc = __builtin_amdgcn_wmma_f32_16x16x32_f16(
                          false, a_lo[b], false, b_hi, (short)0, acc, false, false);
                acc = __builtin_amdgcn_wmma_f32_16x16x32_f16(
                          false, a_hi[b], false, b_lo, (short)0, acc, false, false);
            }
            #pragma unroll
            for (int j = 0; j < 8; ++j) {
                float dist = wq[ktL] - 2.0f * acc[j]; // ||z||^2 const per row
                if (dist < minv[j]) { minv[j] = dist; mini[j] = code; }
            }
        }
        __syncthreads();                 // all waves done reading buf c&1
    }

    // ---- argmin reduce across the 16 lanes holding different codes --------
    #pragma unroll
    for (int m = 1; m < 16; m <<= 1) {
        #pragma unroll
        for (int j = 0; j < 8; ++j) {
            float ov = __shfl_xor(minv[j], m, 32);
            int   oi = __shfl_xor(mini[j], m, 32);
            if (ov < minv[j] || (ov == minv[j] && oi < mini[j])) {
                minv[j] = ov; mini[j] = oi;
            }
        }
    }
    if (col == 0) {
        #pragma unroll
        for (int j = 0; j < 8; ++j)           // C/D layout: M = j + 8*half
            idxOut[tokBase + half * 8 + j] = mini[j];
    }
}

// Gather codebook rows, accumulate loss, counts and per-code embedding sums.
__global__ void __launch_bounds__(256)
k_scatter(const float* __restrict__ z, const float* __restrict__ w,
          const int* __restrict__ idx, float* __restrict__ quant,
          float* __restrict__ n_batch, float* __restrict__ emb_sum,
          float* __restrict__ loss_acc) {
    const int d   = threadIdx.x & 63;
    const int sub = threadIdx.x >> 6;
    float lacc = 0.0f;
    for (int t = blockIdx.x * 4 + sub; t < NTOK; t += gridDim.x * 4) {
        int   code = idx[t];
        float zz   = z[(size_t)t * DDIM + d];
        float q    = w[(size_t)code * DDIM + d];
        quant[(size_t)t * DDIM + d] = q;    // quantized_st == quantized (values)
        float diff = q - zz;
        lacc += diff * diff;
        atomicAdd(&emb_sum[(size_t)code * DDIM + d], zz);
        if (d == 0) atomicAdd(&n_batch[code], 1.0f);
    }
    __shared__ float red[256];
    red[threadIdx.x] = lacc;
    __syncthreads();
    for (int s = 128; s > 0; s >>= 1) {
        if (threadIdx.x < s) red[threadIdx.x] += red[threadIdx.x + s];
        __syncthreads();
    }
    if (threadIdx.x == 0) atomicAdd(loss_acc, red[0]);
}

// EMA updates + normalized codebook.
__global__ void k_finalize(const float* __restrict__ ema_cs,
                           const float* __restrict__ ema_es,
                           const float* __restrict__ n_batch,
                           const float* __restrict__ emb_sum,
                           float* __restrict__ out_ncs,
                           float* __restrict__ out_nes,
                           float* __restrict__ out_nw) {
    int i = blockIdx.x * blockDim.x + threadIdx.x;   // 0 .. K*D-1
    if (i >= KCODE * DDIM) return;
    int k = i >> 6, d = i & 63;
    float ncs = ema_cs[k] * DECAYC + ONE_M_DECAY * n_batch[k];
    if (d == 0) out_ncs[k] = ncs;
    float nes = ema_es[i] * DECAYC + ONE_M_DECAY * emb_sum[i];
    out_nes[i] = nes;
    out_nw[i]  = nes / (ncs + EPSC);
}

// Scalars: commitment loss + perplexity.
__global__ void k_scalar(const float* __restrict__ n_batch,
                         const float* __restrict__ loss_acc,
                         float* __restrict__ out_loss,
                         float* __restrict__ out_perp) {
    float s = 0.0f;
    for (int k = threadIdx.x; k < KCODE; k += 256) {
        float p = n_batch[k] * (1.0f / (float)NTOK);
        s += p * logf(p + 1e-10f);
    }
    __shared__ float red[256];
    red[threadIdx.x] = s;
    __syncthreads();
    for (int st = 128; st > 0; st >>= 1) {
        if (threadIdx.x < st) red[threadIdx.x] += red[threadIdx.x + st];
        __syncthreads();
    }
    if (threadIdx.x == 0) {
        *out_perp = expf(-red[0]);
        *out_loss = COMMIT * (*loss_acc) * (1.0f / (float)(NTOK * DDIM));
    }
}

// ---------------------------------------------------------------------------
extern "C" void kernel_launch(void* const* d_in, const int* in_sizes, int n_in,
                              void* d_out, int out_size, void* d_ws, size_t ws_size,
                              hipStream_t stream) {
    const float* z       = (const float*)d_in[0];   // (B,T,D)
    const float* w       = (const float*)d_in[1];   // (K,D)
    const float* ema_cs  = (const float*)d_in[2];   // (K,)
    const float* ema_es  = (const float*)d_in[3];   // (K,D)

    char* ws = (char*)d_ws;
    int*   idx      = (int*)  (ws + WS_IDX);
    float* n_batch  = (float*)(ws + WS_NBATCH);
    float* emb_sum  = (float*)(ws + WS_EMBSUM);
    float* loss_acc = (float*)(ws + WS_LOSS);
    f16_t* whi      = (f16_t*)(ws + WS_WHI);
    f16_t* wlo      = (f16_t*)(ws + WS_WLO);
    float* wsq      = (float*)(ws + WS_WSQ);

    float* out      = (float*)d_out;
    float* out_q    = out + OUT_QUANT;
    float* out_loss = out + OUT_LOSS;
    float* out_perp = out + OUT_PERP;
    float* out_ncs  = out + OUT_NCS;
    float* out_nes  = out + OUT_NES;
    float* out_nw   = out + OUT_NW;

    // zero n_batch + emb_sum + loss_acc (contiguous region)
    const int nzero = (4096 + 262144 + 256) / 4;
    k_zero<<<(nzero + 255) / 256, 256, 0, stream>>>(n_batch, nzero);

    // pack weight to f16 hi/lo + ||w||^2
    k_prep<<<KCODE, DDIM, 0, stream>>>(w, whi, wlo, wsq);

    // WMMA nearest-code search: 512 blocks x 8 waves x 16 tokens
    k_argmin<<<NTOK / 128, 256, 0, stream>>>(z, whi, wlo, wsq, idx);

    // gather + EMA statistics + loss accumulation
    k_scatter<<<2048, 256, 0, stream>>>(z, w, idx, out_q, n_batch, emb_sum,
                                        loss_acc);

    // EMA update + normalized weight
    k_finalize<<<(KCODE * DDIM + 255) / 256, 256, 0, stream>>>(
        ema_cs, ema_es, n_batch, emb_sum, out_ncs, out_nes, out_nw);

    // scalars
    k_scalar<<<1, 256, 0, stream>>>(n_batch, loss_acc, out_loss, out_perp);
}
